// LigerLMHeadCPO_65970697666952
// MI455X (gfx1250) — matrix operand order repaired and checked by reading, not compiled, not run on hardware
//
#include <hip/hip_runtime.h>
#include <math.h>

// ---------------- problem constants (from reference) ----------------
#define B2   4
#define TT   1024
#define HH   2048
#define VV   32000
#define MM   (B2*TT)           // 4096 rows
#define BM   32                // rows per workgroup
#define BN   256               // cols per workgroup (N chunk)
#define TK   32                // K per step (bf16 WMMA K)
#define NCH  (VV/BN)           // 125 N chunks
#define IGNORE_IDX (-100LL)

typedef __attribute__((ext_vector_type(16))) __bf16        v16bf;
typedef __attribute__((ext_vector_type(8)))  float         v8f;
typedef __attribute__((ext_vector_type(8)))  unsigned int  v8u;
typedef __attribute__((ext_vector_type(4)))  unsigned int  v4u;
typedef __attribute__((ext_vector_type(8)))  int           v8i;
typedef __attribute__((ext_vector_type(4)))  int           v4i;

// order-preserving float <-> uint encoding for atomicMax on floats
__device__ __forceinline__ unsigned int f32_order(float f) {
    unsigned int u = __float_as_uint(f);
    return (u & 0x80000000u) ? ~u : (u | 0x80000000u);
}
__device__ __forceinline__ float order_f32(unsigned int e) {
    return __uint_as_float((e & 0x80000000u) ? (e ^ 0x80000000u) : ~e);
}

// ---- TDM: DMA one 2D tile (tile_dim1 rows x TK bf16) global -> LDS ----
// Tensor: row-major bf16 matrix with row stride HH elements.
// D# layout per CDNA5 ISA ch.8 (groups 0/1; groups 2/3 zero for 2D).
__device__ __forceinline__ void tdm_load_tile_b16(const unsigned short* gsrc,
                                                  unsigned lds_byte_off,
                                                  unsigned rows) {
    unsigned long long ga = (unsigned long long)(uintptr_t)gsrc;
    v4u g0;
    g0[0] = 1u;                                      // count=1 (valid), user mode
    g0[1] = lds_byte_off;                            // LDS dest byte address
    g0[2] = (unsigned)(ga & 0xFFFFFFFFu);            // global_addr[31:0]
    g0[3] = (unsigned)((ga >> 32) & 0x1FFFFFFu)      // global_addr[56:32]
          | (2u << 30);                              // type = 2 ("image")
    v8i g1;
    g1[0] = (int)(1u << 16);                         // wg_mask=0, data_size=1 (2B)
    g1[1] = (int)(((unsigned)HH & 0xFFFFu) << 16);   // tensor_dim0[15:0] @bits[31:16]
    g1[2] = (int)((((unsigned)HH >> 16) & 0xFFFFu)   // tensor_dim0[31:16]
          | (((unsigned)VV & 0xFFFFu) << 16));       // tensor_dim1[15:0]
    g1[3] = (int)((((unsigned)VV >> 16) & 0xFFFFu)   // tensor_dim1[31:16]
          | ((unsigned)TK << 16));                   // tile_dim0 = 32 elements
    g1[4] = (int)rows;                               // tile_dim1 ; tile_dim2=0
    g1[5] = (int)HH;                                 // tensor_dim0_stride[31:0]
    g1[6] = 0;                                       // stride hi / dim1_stride lo
    g1[7] = 0;
    v4i z4 = {0, 0, 0, 0};
#if defined(__clang_major__) && (__clang_major__ >= 23)
    v8i z8 = {0, 0, 0, 0, 0, 0, 0, 0};
    __builtin_amdgcn_tensor_load_to_lds(g0, g1, z4, z4, z8, 0);
#else
    __builtin_amdgcn_tensor_load_to_lds(g0, g1, z4, z4, 0);
#endif
}

// ---------------- 1) f32 -> bf16 (RNE), packed pairs ----------------
__global__ void cvt_f32_bf16_pairs(const float2* __restrict__ in,
                                   unsigned int* __restrict__ out, long n) {
    long i = (long)blockIdx.x * blockDim.x + threadIdx.x;
    long stride = (long)gridDim.x * blockDim.x;
    for (; i < n; i += stride) {
        float2 f = in[i];
        unsigned int a = __float_as_uint(f.x);
        unsigned int b = __float_as_uint(f.y);
        a = (a + 0x7FFFu + ((a >> 16) & 1u)) >> 16;
        b = (b + 0x7FFFu + ((b >> 16) & 1u)) >> 16;
        out[i] = (a & 0xFFFFu) | (b << 16);
    }
}

// ---------------- 2) bf16 WMMA GEMM + online LSE per chunk ----------------
// grid: (MM/BM, NCH) ; block: 256 threads = 8 waves
//  wave wid: wm = wid&1 (two 16-row subtiles), wn = wid>>1 (four 64-col groups)
// B tiles (BN x TK bf16 = 16KB) are TDM-DMA'd into double-buffered LDS.
__global__ __launch_bounds__(256)
void cpo_gemm_lse(const unsigned short* __restrict__ xb,   // [MM][HH] bf16
                  const unsigned short* __restrict__ wb,   // [VV][HH] bf16
                  const long long*      __restrict__ y,    // [MM]
                  float* __restrict__ chunk_max,           // [NCH][MM]
                  float* __restrict__ chunk_sum,           // [NCH][MM]
                  float* __restrict__ label_logit)         // [MM]
{
    __shared__ unsigned short sB[2][BN * TK];   // 2 x 16KB double buffer
    __shared__ unsigned int   s_rowmax[BM];
    __shared__ float          s_rowsum[BM];

    const int tid  = threadIdx.x;
    const int lane = tid & 31;
    const int wid  = tid >> 5;
    const int wm   = wid & 1;
    const int wn   = wid >> 1;
    const int hi   = lane >> 4;      // lane half
    const int lo   = lane & 15;

    const int mbase = blockIdx.x * BM;
    const int nbase = blockIdx.y * BN;
    const int ncolb = nbase + wn * 64;

    if (tid < BM) {
        s_rowmax[tid] = f32_order(-__builtin_inff());
        s_rowsum[tid] = 0.0f;
    }

    // A fragment source: row = mbase + wm*16 + lo ; K runs [hi*16, hi*16+16)
    const unsigned short* aptr = xb + (size_t)(mbase + wm * 16 + lo) * HH + hi * 16;
    // B fragment source in LDS: lane reads column n_local, 16 contiguous K
    const unsigned short* wbase = wb + (size_t)nbase * HH;
    const unsigned lds_off[2] = {
        (unsigned)(unsigned long long)(uintptr_t)(&sB[0][0]),
        (unsigned)(unsigned long long)(uintptr_t)(&sB[1][0])
    };

    v8f acc[4] = {};

    const int KT = HH / TK;          // 64 K-steps

    // prologue: DMA first B tile into buffer 0 (one wave programs the TDM)
    if (wid == 0) {
        tdm_load_tile_b16(wbase, lds_off[0], BN);
    }

    for (int kt = 0; kt < KT; ++kt) {
        const int cur = kt & 1;
        if (wid == 0) {
            if (kt + 1 < KT) {
                tdm_load_tile_b16(wbase + (size_t)(kt + 1) * TK,
                                  lds_off[cur ^ 1], BN);
                __builtin_amdgcn_s_wait_tensorcnt(1);   // current tile landed
            } else {
                __builtin_amdgcn_s_wait_tensorcnt(0);
            }
        }
        __syncthreads();   // publish sB[cur] to all waves

        // A 16x32 bf16 fragment (two b128 global loads per lane)
        const int k0 = kt * TK;
        uint4 a0 = *(const uint4*)(aptr + k0);
        uint4 a1 = *(const uint4*)(aptr + k0 + 8);
        v8u ar; ar[0]=a0.x; ar[1]=a0.y; ar[2]=a0.z; ar[3]=a0.w;
                ar[4]=a1.x; ar[5]=a1.y; ar[6]=a1.z; ar[7]=a1.w;
        v16bf afrag = __builtin_bit_cast(v16bf, ar);
        __builtin_prefetch((const void*)(aptr + k0 + 512), 0, 1);

#pragma unroll
        for (int j = 0; j < 4; ++j) {
            // lane: column n_local = wn*64 + j*16 + lo, K in [hi*16, hi*16+16)
            const unsigned short* bp =
                &sB[cur][(unsigned)(wn * 64 + j * 16 + lo) * TK + hi * 16];
            uint4 b0 = *(const uint4*)(bp);
            uint4 b1 = *(const uint4*)(bp + 8);
            v8u br; br[0]=b0.x; br[1]=b0.y; br[2]=b0.z; br[3]=b0.w;
                    br[4]=b1.x; br[5]=b1.y; br[6]=b1.z; br[7]=b1.w;
            v16bf bfrag = __builtin_bit_cast(v16bf, br);
            acc[j] = __builtin_amdgcn_wmma_f32_16x16x32_bf16(
                false, afrag, false, bfrag, (short)0, acc[j], false, false);
        }
        __syncthreads();   // all reads of sB[cur] done before it is re-filled
    }

    // ---- pass 1: per-row chunk max (+ label-logit scatter) ----
#pragma unroll
    for (int r = 0; r < 8; ++r) {
        const int rowl = wm * 16 + r + hi * 8;       // row within block tile
        const int rowg = mbase + rowl;               // global row
        const long long lab = y[rowg];
        float vmax = -__builtin_inff();
#pragma unroll
        for (int j = 0; j < 4; ++j) {
            float v = acc[j][r];
            vmax = fmaxf(vmax, v);
            int ng = ncolb + j * 16 + lo;
            if ((long long)ng == lab) label_logit[rowg] = v;
        }
        vmax = fmaxf(vmax, __shfl_xor(vmax, 1));
        vmax = fmaxf(vmax, __shfl_xor(vmax, 2));
        vmax = fmaxf(vmax, __shfl_xor(vmax, 4));
        vmax = fmaxf(vmax, __shfl_xor(vmax, 8));
        if (lo == 0) atomicMax(&s_rowmax[rowl], f32_order(vmax));
    }
    __syncthreads();

    // ---- pass 2: per-row sum of exp(v - rowmax) ----
#pragma unroll
    for (int r = 0; r < 8; ++r) {
        const int rowl = wm * 16 + r + hi * 8;
        const float rm = order_f32(s_rowmax[rowl]);
        float s = 0.0f;
#pragma unroll
        for (int j = 0; j < 4; ++j) s += __expf(acc[j][r] - rm);
        s += __shfl_xor(s, 1);
        s += __shfl_xor(s, 2);
        s += __shfl_xor(s, 4);
        s += __shfl_xor(s, 8);
        if (lo == 0) atomicAdd(&s_rowsum[rowl], s);
    }
    __syncthreads();

    if (tid < BM) {
        const int rowg = mbase + tid;
        chunk_max[(size_t)blockIdx.y * MM + rowg] = order_f32(s_rowmax[tid]);
        chunk_sum[(size_t)blockIdx.y * MM + rowg] = s_rowsum[tid];
    }
}

// ---------------- 3) merge chunks -> per-token logprob ----------------
__global__ void cpo_finalize(const float* __restrict__ chunk_max,
                             const float* __restrict__ chunk_sum,
                             const float* __restrict__ label_logit,
                             const long long* __restrict__ y,
                             float* __restrict__ per_tok)
{
    int m = blockIdx.x * blockDim.x + threadIdx.x;
    if (m >= MM) return;
    float gm = -__builtin_inff();
    float gs = 0.0f;
    for (int c = 0; c < NCH; ++c) {
        float mx = chunk_max[(size_t)c * MM + m];
        float s  = chunk_sum[(size_t)c * MM + m];
        if (mx > gm) {
            gs = gs * __expf(gm - mx) + s;
            gm = mx;
        } else {
            gs += s * __expf(mx - gm);
        }
    }
    float lse = gm + logf(gs);
    long long lab = y[m];
    per_tok[m] = (lab != IGNORE_IDX) ? (label_logit[m] - lse) : 0.0f;
}

// ---------------- 4) final scalar loss ----------------
__global__ void cpo_loss(const float* __restrict__ per_tok,
                         const long long* __restrict__ y,
                         float* __restrict__ out)
{
    __shared__ float ssum[B2];
    __shared__ float scnt[B2];
    int t = threadIdx.x;                     // 1024 threads, one per token pos
    if (t < B2) { ssum[t] = 0.0f; scnt[t] = 0.0f; }
    __syncthreads();
#pragma unroll
    for (int b = 0; b < B2; ++b) {
        long long lab = y[(size_t)b * TT + t];
        float v = per_tok[(size_t)b * TT + t];
        bool msk = (lab != IGNORE_IDX);
        atomicAdd(&ssum[b], msk ? v : 0.0f);
        atomicAdd(&scnt[b], msk ? 1.0f : 0.0f);
    }
    __syncthreads();
    if (t == 0) {
        float seq[B2];
#pragma unroll
        for (int b = 0; b < B2; ++b) seq[b] = ssum[b] / fmaxf(scnt[b], 1.0f);
        const float beta = 0.1f, alpha = 1.0f;
        float pref = 0.0f;
#pragma unroll
        for (int i = 0; i < B2 / 2; ++i) {
            float z = beta * (seq[i] - seq[i + B2 / 2]);
            // -log(sigmoid(z)) = softplus(-z) = max(-z,0) + log1p(exp(-|z|))
            pref += fmaxf(-z, 0.0f) + log1pf(expf(-fabsf(z)));
        }
        pref /= (float)(B2 / 2);
        float nll = -(ssum[0] + ssum[1]) / fmaxf(scnt[0] + scnt[1], 1.0f);
        out[0] = alpha * nll + pref;
    }
}

// ---------------- launcher ----------------
extern "C" void kernel_launch(void* const* d_in, const int* in_sizes, int n_in,
                              void* d_out, int out_size, void* d_ws, size_t ws_size,
                              hipStream_t stream) {
    const float*     x = (const float*)d_in[0];      // [B2][TT][HH]
    const float*     W = (const float*)d_in[1];      // [VV][HH]
    const long long* y = (const long long*)d_in[2];  // [B2][TT]
    float* out = (float*)d_out;

    char* ws = (char*)d_ws;
    size_t off = 0;
    unsigned int* xb_u = (unsigned int*)(ws + off); off += (size_t)MM * HH * 2;     // bf16 x
    unsigned int* wb_u = (unsigned int*)(ws + off); off += (size_t)VV * HH * 2;     // bf16 W
    float* chunk_max   = (float*)(ws + off);        off += (size_t)NCH * MM * 4;
    float* chunk_sum   = (float*)(ws + off);        off += (size_t)NCH * MM * 4;
    float* label_logit = (float*)(ws + off);        off += (size_t)MM * 4;
    float* per_tok     = (float*)(ws + off);        off += (size_t)MM * 4;

    // 1) convert x and W to bf16 (packed pairs)
    {
        long npx = (long)MM * HH / 2;
        cvt_f32_bf16_pairs<<<2048, 256, 0, stream>>>((const float2*)x, xb_u, npx);
        long npw = (long)VV * HH / 2;
        cvt_f32_bf16_pairs<<<8192, 256, 0, stream>>>((const float2*)W, wb_u, npw);
    }

    // 2) WMMA GEMM (TDM-fed LDS double buffer) + per-chunk online log-sum-exp
    {
        dim3 grid(MM / BM, NCH);
        cpo_gemm_lse<<<grid, 256, 0, stream>>>(
            (const unsigned short*)xb_u, (const unsigned short*)wb_u, y,
            chunk_max, chunk_sum, label_logit);
    }

    // 3) merge chunks into per-token log-probs
    cpo_finalize<<<MM / 256, 256, 0, stream>>>(chunk_max, chunk_sum, label_logit, y, per_tok);

    // 4) scalar CPO loss
    cpo_loss<<<1, TT, 0, stream>>>(per_tok, y, out);
}
